// SimpleLSTMModel_17729624998516
// MI455X (gfx1250) — matrix operand order repaired and checked by reading, not compiled, not run on hardware
//
#include <hip/hip_runtime.h>
#include <hip/hip_bf16.h>

// ---- problem dims ----
#define B_ 1024
#define T_ 256
#define F_ 256
#define U_ 512
#define D_ 256
#define O_ 8
#define G_ (4 * U_) // 2048 gate width

typedef __attribute__((ext_vector_type(16))) __bf16 bf16x16;
typedef __attribute__((ext_vector_type(8)))  __bf16 bf16x8;
typedef __attribute__((ext_vector_type(8)))  float  f32x8;
typedef __attribute__((ext_vector_type(4)))  unsigned int u32x4;
typedef __attribute__((vector_size(4 * sizeof(int)))) int i32x4v; // builtin param type

// ---- CDNA5 async global->LDS copy (bypasses VGPRs, tracked on ASYNCcnt) ----
#if defined(__has_builtin)
#if __has_builtin(__builtin_amdgcn_global_load_async_to_lds_b128)
#define HAVE_ASYNC_LDS 1
#endif
#if __has_builtin(__builtin_amdgcn_s_wait_asynccnt)
#define HAVE_WAIT_ASYNC 1
#endif
#endif
#ifndef HAVE_ASYNC_LDS
#define HAVE_ASYNC_LDS 0
#endif
#ifndef HAVE_WAIT_ASYNC
#define HAVE_WAIT_ASYNC 0
#endif

__device__ __forceinline__ void async_copy_b128(const __bf16* gp, __bf16* lp) {
#if HAVE_ASYNC_LDS
    __builtin_amdgcn_global_load_async_to_lds_b128(
        (i32x4v*)gp, (i32x4v*)lp, 0, 0);
#else
    *(u32x4*)lp = *(const u32x4*)gp;
#endif
}

__device__ __forceinline__ void async_wait0() {
#if HAVE_ASYNC_LDS
#if HAVE_WAIT_ASYNC
    __builtin_amdgcn_s_wait_asynccnt(0);
#else
    asm volatile("s_wait_asynccnt 0x0" ::: "memory");
#endif
#endif
}

__device__ __forceinline__ bf16x16 frag16(const __bf16* p0, const __bf16* p1) {
    bf16x8 c0 = *(const bf16x8*)p0;
    bf16x8 c1 = *(const bf16x8*)p1;
    return __builtin_shufflevector(c0, c1, 0, 1, 2, 3, 4, 5, 6, 7,
                                   8, 9, 10, 11, 12, 13, 14, 15);
}

__device__ __forceinline__ float sigmoidf_(float x) {
    return 1.f / (1.f + __expf(-x));
}

// ------------------------------------------------------------------
// fp32 -> bf16 conversion (plain)
// ------------------------------------------------------------------
__global__ void cvt_f32_bf16(const float* __restrict__ src,
                             __bf16* __restrict__ dst, int n) {
    int i = blockIdx.x * blockDim.x + threadIdx.x;
    if (i < n) dst[i] = (__bf16)src[i];
}

// ------------------------------------------------------------------
// fp32 -> bf16 conversion with transpose: src[K][N] -> dst[N][K].
// One-time cost so in-loop B staging becomes a straight block copy.
// ------------------------------------------------------------------
__global__ void cvt_f32_bf16_t(const float* __restrict__ src,
                               __bf16* __restrict__ dst, int K, int N) {
    int i = blockIdx.x * blockDim.x + threadIdx.x;
    if (i < K * N) {
        int k = i / N;
        int n = i - k * N;
        dst[(long long)n * K + k] = (__bf16)src[i];
    }
}

// ------------------------------------------------------------------
// zero-init of LSTM state (cell fp32 + h bf16)
// ------------------------------------------------------------------
__global__ void zero_state(float* __restrict__ c, __bf16* __restrict__ hbf,
                           int n) {
    int i = blockIdx.x * blockDim.x + threadIdx.x;
    if (i < n) { c[i] = 0.f; hbf[i] = (__bf16)0.f; }
}

// ------------------------------------------------------------------
// Generic 128x128x32-tile WMMA bf16 GEMM, fp32 accumulate.
// A: [M][K] row-major bf16.  Bt: [N][K] (pre-transposed weights).
// 256 threads = 8 waves (wave32); wave grid 2(M) x 4(N); each wave
// computes a 64x32 tile via 4x2 v_wmma_f32_16x16x32_bf16 accumulators.
// Double-buffered LDS; BOTH tiles staged with async global->LDS b128.
// MODE 0: out_bf16 = A@B + bias          (input projection xg)
// MODE 2: out_f32  = relu(A@B + bias)    (MLP layer 1)
// ------------------------------------------------------------------
template <int MODE>
__global__ __launch_bounds__(256) void wmma_gemm(
    const __bf16* __restrict__ A, int lda,
    const __bf16* __restrict__ Bt, int ldbt,
    float* __restrict__ outF, __bf16* __restrict__ outH, int ldc,
    const float* __restrict__ bias, int K)
{
    __shared__ __align__(16) __bf16 As[2][128][40];
    __shared__ __align__(16) __bf16 Bs[2][128][40]; // Bs[.][n][k]

    const int tid  = threadIdx.x;
    const int wave = tid >> 5;
    const int lane = tid & 31;
    const int lm = lane & 15;
    const int lh = lane >> 4;
    const int wm = wave & 1;       // wave M coordinate (0..1)
    const int wn = wave >> 1;      // wave N coordinate (0..3)

    const long long blockM = (long long)blockIdx.y * 128;
    const int       blockN = blockIdx.x * 128;

    // both tiles are 128 rows x 32 cols: 2 x b128 async copies per thread
    const int arow = tid >> 2;           // 0..63
    const int acol = (tid & 3) * 8;      // 0,8,16,24

    auto stageA = [&](int buf, int k0) {
        async_copy_b128(A + (blockM + arow) * (long long)lda + k0 + acol,
                        &As[buf][arow][acol]);
        async_copy_b128(A + (blockM + arow + 64) * (long long)lda + k0 + acol,
                        &As[buf][arow + 64][acol]);
    };
    auto stageB = [&](int buf, int k0) {
        async_copy_b128(Bt + (long long)(blockN + arow) * ldbt + k0 + acol,
                        &Bs[buf][arow][acol]);
        async_copy_b128(Bt + (long long)(blockN + arow + 64) * ldbt + k0 + acol,
                        &Bs[buf][arow + 64][acol]);
    };

    const f32x8 vzero = {0.f, 0.f, 0.f, 0.f, 0.f, 0.f, 0.f, 0.f};
    f32x8 acc[4][2];
#pragma unroll
    for (int i = 0; i < 4; ++i)
#pragma unroll
        for (int j = 0; j < 2; ++j) acc[i][j] = vzero;

    stageA(0, 0);
    stageB(0, 0);

    int buf = 0;
    for (int k0 = 0; k0 < K; k0 += 32) {
        async_wait0();
        __syncthreads();

        const int kn = k0 + 32;
        if (kn < K) {
            stageA(buf ^ 1, kn);
            stageB(buf ^ 1, kn);
        }
        if (kn + 32 < K) { // warm L2 one double-buffer slot ahead
            __builtin_prefetch(A + (blockM + arow) * (long long)lda + kn + 32 + acol, 0, 1);
            __builtin_prefetch(Bt + (long long)(blockN + arow) * ldbt + kn + 32 + acol, 0, 1);
        }

        bf16x16 afrag[4], bfrag[2];
#pragma unroll
        for (int i = 0; i < 4; ++i) {
            const int m = wm * 64 + i * 16 + lm;
            afrag[i] = frag16(&As[buf][m][16 * lh], &As[buf][m][16 * lh + 8]);
        }
#pragma unroll
        for (int j = 0; j < 2; ++j) {
            const int n = wn * 32 + j * 16 + lm;
            bfrag[j] = frag16(&Bs[buf][n][16 * lh], &Bs[buf][n][16 * lh + 8]);
        }
#pragma unroll
        for (int i = 0; i < 4; ++i)
#pragma unroll
            for (int j = 0; j < 2; ++j)
                acc[i][j] = __builtin_amdgcn_wmma_f32_16x16x32_bf16(
                    false, afrag[i], false, bfrag[j], (short)0, acc[i][j],
                    false, false);
        buf ^= 1;
    }

#pragma unroll
    for (int i = 0; i < 4; ++i) {
#pragma unroll
        for (int j = 0; j < 2; ++j) {
#pragma unroll
            for (int r = 0; r < 8; ++r) {
                const long long m = blockM + wm * 64 + i * 16 + lh * 8 + r;
                const int n = blockN + wn * 32 + j * 16 + lm;
                float v = acc[i][j][r] + bias[n];
                if (MODE == 0) {
                    outH[m * ldc + n] = (__bf16)v;
                } else {
                    outF[m * ldc + n] = v > 0.f ? v : 0.f;
                }
            }
        }
    }
}

// ------------------------------------------------------------------
// Fused LSTM step: one kernel per timestep.
// gates[:, {u, U+u, 2U+u, 3U+u}] computed in-block so the nonlinear
// cell update happens in the GEMM epilogue (no gates round-trip).
// Block = 128 rows(b) x 64 units(u) -> 256 N columns (4 gate groups).
// 8 waves: 2(M) x 4(u); each wave: 4 M-subtiles x 4 gate accumulators
// = 16 v8f accumulators, all mapping to the same (m,u) per element.
// rect: [4U][U] pre-transposed recurrent weights (bf16).
//   gates = hIn @ rec + xg[:,t,:]
//   c = sig(f)*c + sig(i)*relu(g);  hOut = sig(o)*relu(c)
// ------------------------------------------------------------------
__global__ __launch_bounds__(256) void lstm_step_fused(
    const __bf16* __restrict__ hIn,   // [B, U] bf16
    const __bf16* __restrict__ rect,  // [4U, U] bf16 (transposed rec)
    const __bf16* __restrict__ xgT,   // xg + t*G, row stride T*G
    float* __restrict__ cSt,          // [B, U] fp32 (in/out)
    __bf16* __restrict__ hOut)        // [B, U] bf16
{
    __shared__ __align__(16) __bf16 As[2][128][40];
    __shared__ __align__(16) __bf16 Bs[2][256][40]; // Bs[.][64*gate+du][k]

    const int tid  = threadIdx.x;
    const int wave = tid >> 5;
    const int lane = tid & 31;
    const int lm = lane & 15;
    const int lh = lane >> 4;
    const int wm = wave & 1;       // M coordinate (0..1)
    const int wu = wave >> 1;      // u coordinate (0..3)

    const long long blockM = (long long)blockIdx.y * 128;
    const int       u0     = blockIdx.x * 64;

    const int arow = tid >> 2;           // 0..63
    const int acol = (tid & 3) * 8;      // 0,8,16,24
    // B staging: 256 rows of rect (4 gate groups of 64), 32 cols/step;
    // thread owns one LDS row: 4 async b128 copies.
    const int brow = tid;                                   // local row
    const int bgrow = 512 * (tid >> 6) + u0 + (tid & 63);   // rect row

    auto stageA = [&](int buf, int k0) {
        async_copy_b128(hIn + (blockM + arow) * (long long)U_ + k0 + acol,
                        &As[buf][arow][acol]);
        async_copy_b128(hIn + (blockM + arow + 64) * (long long)U_ + k0 + acol,
                        &As[buf][arow + 64][acol]);
    };
    auto stageB = [&](int buf, int k0) {
        const __bf16* g = rect + (long long)bgrow * U_ + k0;
#pragma unroll
        for (int c = 0; c < 32; c += 8)
            async_copy_b128(g + c, &Bs[buf][brow][c]);
    };

    const f32x8 vzero = {0.f, 0.f, 0.f, 0.f, 0.f, 0.f, 0.f, 0.f};
    f32x8 acc[4][4]; // [M-subtile][gate]
#pragma unroll
    for (int i = 0; i < 4; ++i)
#pragma unroll
        for (int g = 0; g < 4; ++g) acc[i][g] = vzero;

    stageA(0, 0);
    stageB(0, 0);

    int buf = 0;
    for (int k0 = 0; k0 < U_; k0 += 32) {
        async_wait0();
        __syncthreads();

        const int kn = k0 + 32;
        if (kn < U_) {
            stageA(buf ^ 1, kn);
            stageB(buf ^ 1, kn);
        }

        bf16x16 afrag[4], bfrag[4];
#pragma unroll
        for (int i = 0; i < 4; ++i) {
            const int m = wm * 64 + i * 16 + lm;
            afrag[i] = frag16(&As[buf][m][16 * lh], &As[buf][m][16 * lh + 8]);
        }
#pragma unroll
        for (int g = 0; g < 4; ++g) {
            const int n = g * 64 + wu * 16 + lm;
            bfrag[g] = frag16(&Bs[buf][n][16 * lh], &Bs[buf][n][16 * lh + 8]);
        }
#pragma unroll
        for (int i = 0; i < 4; ++i)
#pragma unroll
            for (int g = 0; g < 4; ++g)
                acc[i][g] = __builtin_amdgcn_wmma_f32_16x16x32_bf16(
                    false, afrag[i], false, bfrag[g], (short)0, acc[i][g],
                    false, false);
        buf ^= 1;
    }

    // ---- fused epilogue: gate nonlinearity + cell update ----
    const int u = u0 + wu * 16 + lm;
#pragma unroll
    for (int i = 0; i < 4; ++i) {
#pragma unroll
        for (int r = 0; r < 8; ++r) {
            const long long m = blockM + wm * 64 + i * 16 + lh * 8 + r;
            const __bf16* xr = xgT + m * ((long long)T_ * G_);
            const float gi = acc[i][0][r] + (float)xr[0 * U_ + u];
            const float gf = acc[i][1][r] + (float)xr[1 * U_ + u];
            const float gg = acc[i][2][r] + (float)xr[2 * U_ + u];
            const float go = acc[i][3][r] + (float)xr[3 * U_ + u];
            const long long su = m * U_ + u;
            const float cOld = cSt[su];
            const float rg = gg > 0.f ? gg : 0.f;
            const float cn = sigmoidf_(gf) * cOld + sigmoidf_(gi) * rg;
            const float hn = sigmoidf_(go) * (cn > 0.f ? cn : 0.f);
            cSt[su] = cn;
            hOut[su] = (__bf16)hn;
        }
    }
}

// ------------------------------------------------------------------
// Final tiny layer: out[b,o] = a1[b,:] . w2[:,o] + b2[o]   (N=8)
// ------------------------------------------------------------------
__global__ void mlp2_kernel(const float* __restrict__ a1,
                            const float* __restrict__ w2,
                            const float* __restrict__ b2,
                            float* __restrict__ out) {
    const int idx = blockIdx.x * blockDim.x + threadIdx.x; // B_*O_
    const int b = idx >> 3;
    const int o = idx & 7;
    const float* row = a1 + (long long)b * D_;
    float s = b2[o];
#pragma unroll 8
    for (int k = 0; k < D_; ++k) s += row[k] * w2[k * O_ + o];
    out[idx] = s;
}

// ------------------------------------------------------------------
// Host-side orchestration
// ------------------------------------------------------------------
extern "C" void kernel_launch(void* const* d_in, const int* in_sizes, int n_in,
                              void* d_out, int out_size, void* d_ws,
                              size_t ws_size, hipStream_t stream) {
    const float* x      = (const float*)d_in[0];
    const float* kernel = (const float*)d_in[1];
    const float* rec    = (const float*)d_in[2];
    const float* bias   = (const float*)d_in[3];
    const float* w1     = (const float*)d_in[4];
    const float* b1     = (const float*)d_in[5];
    const float* w2     = (const float*)d_in[6];
    const float* b2     = (const float*)d_in[7];
    float* out = (float*)d_out;

    // workspace carve-out (256B aligned)
    char* wp = (char*)d_ws;
    auto carve = [&](size_t bytes) {
        char* p = wp;
        wp += (bytes + 255) & ~(size_t)255;
        return p;
    };
    __bf16* xBf   = (__bf16*)carve((size_t)B_ * T_ * F_ * 2);  // 134 MB
    __bf16* kT    = (__bf16*)carve((size_t)F_ * G_ * 2);       // [G][F], 1 MB
    __bf16* rT    = (__bf16*)carve((size_t)U_ * G_ * 2);       // [G][U], 2 MB
    __bf16* w1T   = (__bf16*)carve((size_t)U_ * D_ * 2);       // [D][U], 0.25 MB
    __bf16* xg    = (__bf16*)carve((size_t)B_ * T_ * G_ * 2);  // 1 GB
    float*  cSt   = (float*)carve((size_t)B_ * U_ * 4);        // 2 MB
    __bf16* hBf0  = (__bf16*)carve((size_t)B_ * U_ * 2);       // 1 MB
    __bf16* hBf1  = (__bf16*)carve((size_t)B_ * U_ * 2);       // 1 MB
    float*  a1    = (float*)carve((size_t)B_ * D_ * 4);        // 1 MB

    // 1) convert operands to bf16 (weights transposed to [N][K])
    {
        int n = B_ * T_ * F_;
        cvt_f32_bf16<<<n / 256, 256, 0, stream>>>(x, xBf, n);
        n = F_ * G_;
        cvt_f32_bf16_t<<<n / 256, 256, 0, stream>>>(kernel, kT, F_, G_);
        n = U_ * G_;
        cvt_f32_bf16_t<<<n / 256, 256, 0, stream>>>(rec, rT, U_, G_);
        n = U_ * D_;
        cvt_f32_bf16_t<<<n / 256, 256, 0, stream>>>(w1, w1T, U_, D_);
    }

    // 2) input projection: xg[B*T, 4U] = x @ kernel + bias  (bf16 out)
    {
        dim3 grid(G_ / 128, (B_ * T_) / 128); // 16 x 2048
        wmma_gemm<0><<<grid, 256, 0, stream>>>(
            xBf, F_, kT, F_, nullptr, xg, G_, bias, F_);
    }

    // 3) init h, c
    {
        int n = B_ * U_;
        zero_state<<<n / 256, 256, 0, stream>>>(cSt, hBf0, n);
    }

    // 4) recurrence: 256 fused steps (GEMM + gate update in one kernel)
    __bf16* hbufs[2] = {hBf0, hBf1};
    for (int t = 0; t < T_; ++t) {
        dim3 grid(U_ / 64, B_ / 128); // 8 x 8
        lstm_step_fused<<<grid, 256, 0, stream>>>(
            hbufs[t & 1], rT, xg + (long long)t * G_, cSt, hbufs[(t + 1) & 1]);
    }
    __bf16* hFinal = hbufs[T_ & 1]; // T_=256 even -> hBf0

    // 5) MLP layer 1: a1 = relu(h @ w1 + b1)
    {
        dim3 grid(D_ / 128, B_ / 128); // 2 x 8
        wmma_gemm<2><<<grid, 256, 0, stream>>>(
            hFinal, U_, w1T, U_, a1, nullptr, D_, b1, U_);
    }

    // 6) final tiny layer
    mlp2_kernel<<<(B_ * O_) / 256, 256, 0, stream>>>(a1, w2, b2, out);
}
